// JanusCrossAttention_45526653337854
// MI455X (gfx1250) — matrix operand order repaired and checked by reading, not compiled, not run on hardware
//
#include <hip/hip_runtime.h>
#include <hip/hip_bf16.h>
#include <stdint.h>

// ---------------------------------------------------------------------------
// JanusCrossAttention for MI455X (gfx1250, wave32, WMMA + async-to-LDS).
// Pipeline: [GEMM wq/wk/wv bf16-WMMA] -> [per-head RMSNorm -> bf16]
//           -> [causal flash-attention, bf16 WMMA, fp32 online softmax,
//               K/V tiles staged via GLOBAL_LOAD_ASYNC_TO_LDS_B128]
//           -> [GEMM wo bf16-WMMA -> fp32 out, A tile staged async]
// ---------------------------------------------------------------------------

typedef __attribute__((ext_vector_type(16))) __bf16 v16bf;
typedef __attribute__((ext_vector_type(8)))  __bf16 v8bf;
typedef __attribute__((ext_vector_type(8)))  float  v8f;

union Frag16 { v16bf v; struct { v8bf lo, hi; } h; };

__device__ __forceinline__ unsigned short f32_to_bf16_us(float f) {
  union { float f; uint32_t u; } cv; cv.f = f;
  uint32_t u = cv.u;
  uint32_t r = u + 0x7FFFu + ((u >> 16) & 1u);   // round-to-nearest-even
  return (unsigned short)(r >> 16);
}
__device__ __forceinline__ unsigned short to_bf16_us(float f)          { return f32_to_bf16_us(f); }
__device__ __forceinline__ unsigned short to_bf16_us(unsigned short u) { return u; }

__device__ __forceinline__ v8f wmma_bf16(v16bf a, v16bf b, v8f c) {
  return __builtin_amdgcn_wmma_f32_16x16x32_bf16(false, a, false, b, (short)0, c,
                                                 false, false);
}

// --- CDNA5 async memory->LDS copy (ASYNCcnt-tracked, no VGPR data path) ----
// vdst VGPR = LDS byte address (addrspace(3) offset = low 32 bits of generic
// shared pointer); vaddr = 64-bit global address; saddr = off.
__device__ __forceinline__ void async_copy_b128(uint32_t lds_addr, const void* gptr) {
  uint64_t ga = (uint64_t)(uintptr_t)gptr;
  asm volatile("global_load_async_to_lds_b128 %0, %1, off"
               :: "v"(lds_addr), "v"(ga)
               : "memory");
}
__device__ __forceinline__ void wait_asynccnt0() {
#if __has_builtin(__builtin_amdgcn_s_wait_asynccnt)
  __builtin_amdgcn_s_wait_asynccnt(0);
#else
  asm volatile("s_wait_asynccnt 0x0" ::: "memory");
#endif
}

// ---------------------------------------------------------------------------
// Tiled GEMM:  C[M,N] (f32) = A[M,K] x Bw[K,N].  A is f32 (converted to bf16
// while staging) or bf16/ushort (staged via async-to-LDS).  Bw is f32 weights
// converted while staging.  Block 256 thr = 8 waves; tile 128x128x32; each
// wave computes 32x64 (2x4 WMMA accumulators).
// ---------------------------------------------------------------------------
template <typename TA>
__global__ __launch_bounds__(256) void gemm_wmma_kernel(
    const TA* __restrict__ A, const float* __restrict__ Bw, float* __restrict__ C,
    int M, int N, int K)
{
  __shared__ __attribute__((aligned(32))) unsigned short As[128][32];   // [m][k]
  __shared__ __attribute__((aligned(32))) unsigned short Bs[128][32];   // [n][k]
  const int tid  = threadIdx.x;
  const int wid  = tid >> 5;
  const int lane = tid & 31;
  const int hl   = lane >> 4;     // half-wave: 0 or 1
  const int ln   = lane & 15;
  const int wm = (wid & 3) * 32;  // wave row offset in tile
  const int wn = (wid >> 2) * 64; // wave col offset in tile
  const int m0 = blockIdx.x * 128;
  const int n0 = blockIdx.y * 128;

  v8f zero = {};
  v8f acc[2][4];
#pragma unroll
  for (int i = 0; i < 2; ++i)
#pragma unroll
    for (int j = 0; j < 4; ++j) acc[i][j] = zero;

  const int nk = K >> 5;
  for (int kt = 0; kt < nk; ++kt) {
    const int k0 = kt << 5;
    __syncthreads();
    // stage A 128x32
    if constexpr (sizeof(TA) == 2) {
      // bf16 source: DMA straight to LDS (2 x b128 per thread)
      int r = tid >> 1, c = (tid & 1) << 4;
      const TA* ap = A + (size_t)(m0 + r) * K + (k0 + c);
      uint32_t la = (uint32_t)(uintptr_t)&As[r][c];
      async_copy_b128(la,      ap);
      async_copy_b128(la + 16, ap + 8);
    } else {
      // f32 source: load + convert through VGPRs
#pragma unroll
      for (int i = 0; i < 16; ++i) {
        int idx = i * 256 + tid;
        int r = idx >> 5, c = idx & 31;
        As[r][c] = to_bf16_us(A[(size_t)(m0 + r) * K + (k0 + c)]);
      }
    }
    // stage B 32x128, stored transposed [n][k] so B-fragments are contiguous
#pragma unroll
    for (int i = 0; i < 16; ++i) {
      int idx = i * 256 + tid;
      int r = idx >> 7, c = idx & 127;   // r = k, c = n
      Bs[c][r] = f32_to_bf16_us(Bw[(size_t)(k0 + r) * N + (n0 + c)]);
    }
    if (kt + 1 < nk) {  // global_prefetch_b8 next tiles
      __builtin_prefetch(&A[(size_t)(m0 + (tid >> 1)) * K + k0 + 32 + ((tid & 1) << 4)], 0, 1);
      __builtin_prefetch(&Bw[(size_t)(k0 + 32 + (tid >> 3)) * N + n0 + ((tid & 7) << 4)], 0, 1);
    }
    if constexpr (sizeof(TA) == 2) wait_asynccnt0();  // own async copies landed
    __syncthreads();                                  // publish LDS to all waves

    v16bf af[2], bfm[4];
#pragma unroll
    for (int ti = 0; ti < 2; ++ti) {
      // A-frag (16x32 bf16): lane m = ln; elems K = hl*16 + e (contiguous)
      const unsigned short* p = &As[wm + ti * 16 + ln][0];
      Frag16 f;
      f.h.lo = *(const v8bf*)(p + hl * 16);
      f.h.hi = *(const v8bf*)(p + hl * 16 + 8);
      af[ti] = f.v;
    }
#pragma unroll
    for (int tj = 0; tj < 4; ++tj)
      // B-frag (32x16 bf16): lane n = ln; elems K = hl*16 + e (contiguous)
      bfm[tj] = *(const v16bf*)&Bs[wn + tj * 16 + ln][hl * 16];
#pragma unroll
    for (int ti = 0; ti < 2; ++ti)
#pragma unroll
      for (int tj = 0; tj < 4; ++tj)
        acc[ti][tj] = wmma_bf16(af[ti], bfm[tj], acc[ti][tj]);
  }

  // C/D layout: VGPR r -> row = r + hl*8, col = ln
#pragma unroll
  for (int ti = 0; ti < 2; ++ti)
#pragma unroll
    for (int tj = 0; tj < 4; ++tj)
#pragma unroll
      for (int r = 0; r < 8; ++r) {
        int row = m0 + wm + ti * 16 + r + hl * 8;
        int col = n0 + wn + tj * 16 + ln;
        C[(size_t)row * N + col] = acc[ti][tj][r];
      }
}

// ---------------------------------------------------------------------------
// Per-head RMSNorm over 128 elems; one wave32 per head-vector; out = bf16.
// ---------------------------------------------------------------------------
__global__ __launch_bounds__(256) void rmsnorm_head_kernel(
    const float* __restrict__ X, const float* __restrict__ w,
    unsigned short* __restrict__ out, int nvec)
{
  int wid = threadIdx.x >> 5, lane = threadIdx.x & 31;
  int vec = blockIdx.x * 8 + wid;
  if (vec >= nvec) return;
  const float4 v = ((const float4*)(X + (size_t)vec * 128))[lane];
  float ss = v.x * v.x + v.y * v.y + v.z * v.z + v.w * v.w;
#pragma unroll
  for (int off = 1; off < 32; off <<= 1) ss += __shfl_xor(ss, off, 32);
  float sc = rsqrtf(ss * (1.0f / 128.0f) + 1e-5f);
  const float4 g = ((const float4*)w)[lane];
  unsigned short* o = out + (size_t)vec * 128 + lane * 4;
  o[0] = f32_to_bf16_us(v.x * sc * g.x);
  o[1] = f32_to_bf16_us(v.y * sc * g.y);
  o[2] = f32_to_bf16_us(v.z * sc * g.z);
  o[3] = f32_to_bf16_us(v.w * sc * g.w);
}

__global__ void cast_bf16_kernel(const float* __restrict__ in,
                                 unsigned short* __restrict__ out, int n)
{
  int i = blockIdx.x * blockDim.x + threadIdx.x;
  int stride = gridDim.x * blockDim.x;
  for (; i < n; i += stride) out[i] = f32_to_bf16_us(in[i]);
}

// ---------------------------------------------------------------------------
// Causal flash attention with GQA (4 q heads share a kv head).
// Grid: (S/64, H, B); block 128 thr = 4 waves; each wave owns 16 query rows.
// K tile and V tile are DMA'd memory->LDS with async-to-LDS; V is then
// transposed LDS->LDS so PV B-fragments are contiguous ds_load_b128 pairs.
// Per 32-key tile: S = Q(16x128)·K^T via 8 WMMA, online softmax in regs,
// O += P(16x32)·V via 8 WMMA.  Q fragments stay resident in VGPRs.
// ---------------------------------------------------------------------------
__global__ __launch_bounds__(128) void flash_attn_kernel(
    const unsigned short* __restrict__ Q, const unsigned short* __restrict__ Kg,
    const unsigned short* __restrict__ Vg, unsigned short* __restrict__ AO)
{
  constexpr int S = 2048, H = 16, Dh = 128, KVW = 512;  // KVW = 4 kv heads * 128
  __shared__ __attribute__((aligned(32))) unsigned short Ks[32][128];    // [key][d]
  __shared__ __attribute__((aligned(32))) unsigned short Vrow[32][128];  // [key][d]
  __shared__ __attribute__((aligned(32))) unsigned short Vs[128][32];    // [d][key]
  __shared__ __attribute__((aligned(32))) unsigned short Ps[4][16][32];  // per-wave P

  const int tid = threadIdx.x;
  const int wid = tid >> 5, lane = tid & 31, hl = lane >> 4, ln = lane & 15;
  const int qt = blockIdx.x, h = blockIdx.y, b = blockIdx.z;
  const int kvh = h >> 2;                      // N_REP = 4
  const int q0w = qt * 64 + wid * 16;          // this wave's first query row
  const float scale = 0.08838834764831845f;    // 1/sqrt(128)

  // resident Q fragments: 16 rows x 128 d = 4 fragments of 16x32
  v16bf qf[4];
  {
    const unsigned short* qptr = Q + ((size_t)(b * S + q0w + ln)) * (H * Dh) + h * Dh;
#pragma unroll
    for (int kd = 0; kd < 4; ++kd) {
      Frag16 f;
      f.h.lo = *(const v8bf*)(qptr + kd * 32 + hl * 16);
      f.h.hi = *(const v8bf*)(qptr + kd * 32 + hl * 16 + 8);
      qf[kd] = f.v;
    }
  }

  v8f zero = {};
  v8f acc[8];
#pragma unroll
  for (int j = 0; j < 8; ++j) acc[j] = zero;
  float mrow[8], lrow[8];
#pragma unroll
  for (int r = 0; r < 8; ++r) { mrow[r] = -3.0e38f; lrow[r] = 0.0f; }

  const int ktmax = qt * 2 + 1;                 // covers keys up to qt*64+63
  for (int kt = 0; kt <= ktmax; ++kt) {
    __syncthreads();   // all waves done reading previous Ks/Vrow/Vs
    { // async DMA: K tile and V tile (each 32 keys x 128 d, 8 KB) -> LDS
      int key = tid >> 2, dseg = (tid & 3) * 32;
      size_t grow = ((size_t)(b * S + kt * 32 + key)) * KVW + kvh * Dh + dseg;
      const unsigned short* kp = Kg + grow;
      const unsigned short* vp = Vg + grow;
      uint32_t lk = (uint32_t)(uintptr_t)&Ks[key][dseg];
      uint32_t lv = (uint32_t)(uintptr_t)&Vrow[key][dseg];
#pragma unroll
      for (int i = 0; i < 4; ++i) {
        async_copy_b128(lk + i * 16, kp + i * 8);
        async_copy_b128(lv + i * 16, vp + i * 8);
      }
    }
    wait_asynccnt0();  // this wave's async copies have landed in LDS
    __syncthreads();   // every wave's copies published

    // transpose V tile LDS->LDS: Vs[d][key] = Vrow[key][d]
    {
      int d = tid;
#pragma unroll
      for (int key = 0; key < 32; ++key)
        Vs[d][key] = Vrow[key][d];
    }

    // scores: two 16x16 tiles (keys 0-15 and 16-31), contract d in 4 steps
    v8f s0 = zero, s1 = zero;
#pragma unroll
    for (int kd = 0; kd < 4; ++kd) {
      v16bf k0f = *(const v16bf*)&Ks[ln][kd * 32 + hl * 16];
      v16bf k1f = *(const v16bf*)&Ks[16 + ln][kd * 32 + hl * 16];
      s0 = wmma_bf16(qf[kd], k0f, s0);
      s1 = wmma_bf16(qf[kd], k1f, s1);
    }

    // online softmax (row = r + hl*8, col = ln); 16-lane xor reductions
#pragma unroll
    for (int r = 0; r < 8; ++r) {
      int qrow = q0w + r + hl * 8;
      int key0 = kt * 32 + ln;
      float v0 = (key0      <= qrow) ? s0[r] * scale : -3.0e38f;
      float v1 = (key0 + 16 <= qrow) ? s1[r] * scale : -3.0e38f;
      float mx = fmaxf(v0, v1);
#pragma unroll
      for (int off = 1; off < 16; off <<= 1) mx = fmaxf(mx, __shfl_xor(mx, off, 32));
      float mnew  = fmaxf(mrow[r], mx);
      float alpha = __expf(mrow[r] - mnew);
      float p0 = __expf(v0 - mnew);
      float p1 = __expf(v1 - mnew);
      float rs = p0 + p1;
#pragma unroll
      for (int off = 1; off < 16; off <<= 1) rs += __shfl_xor(rs, off, 32);
      lrow[r] = lrow[r] * alpha + rs;
      mrow[r] = mnew;
#pragma unroll
      for (int j = 0; j < 8; ++j) acc[j][r] *= alpha;
      Ps[wid][r + hl * 8][ln]      = f32_to_bf16_us(p0);
      Ps[wid][r + hl * 8][16 + ln] = f32_to_bf16_us(p1);
    }
    __syncthreads();   // Vs transpose + P reshape published

    Frag16 pf;
    pf.h.lo = *(const v8bf*)&Ps[wid][ln][hl * 16];
    pf.h.hi = *(const v8bf*)&Ps[wid][ln][hl * 16 + 8];
#pragma unroll
    for (int j = 0; j < 8; ++j) {
      v16bf vbf = *(const v16bf*)&Vs[j * 16 + ln][hl * 16];
      acc[j] = wmma_bf16(pf.v, vbf, acc[j]);
    }
  }

  // normalize and write bf16 output [token][h*128 + d]
#pragma unroll
  for (int r = 0; r < 8; ++r) {
    int qrow = q0w + r + hl * 8;
    float inv = 1.0f / lrow[r];
    unsigned short* o = AO + ((size_t)(b * S + qrow)) * (H * Dh) + h * Dh + ln;
#pragma unroll
    for (int j = 0; j < 8; ++j)
      o[j * 16] = f32_to_bf16_us(acc[j][r] * inv);
  }
}

// ---------------------------------------------------------------------------
extern "C" void kernel_launch(void* const* d_in, const int* in_sizes, int n_in,
                              void* d_out, int out_size, void* d_ws, size_t ws_size,
                              hipStream_t stream) {
  (void)in_sizes; (void)n_in; (void)out_size; (void)ws_size;
  const float* q_stream = (const float*)d_in[0];
  const float* kv_stream = (const float*)d_in[1];
  const float* wq = (const float*)d_in[2];
  const float* wk = (const float*)d_in[3];
  const float* wv = (const float*)d_in[4];
  const float* wo = (const float*)d_in[5];
  const float* q_norm_w = (const float*)d_in[6];
  const float* k_norm_w = (const float*)d_in[7];
  float* out = (float*)d_out;

  const int BS = 2 * 2048;     // B*S token rows
  // workspace carve-up (~92.3 MB)
  float* XQ = (float*)d_ws;                                   // [BS, 2048] f32
  float* XK = XQ + (size_t)BS * 2048;                         // [BS, 512]  f32
  float* XV = XK + (size_t)BS * 512;                          // [BS, 512]  f32
  unsigned short* Qb = (unsigned short*)(XV + (size_t)BS * 512);  // [BS,2048] bf16
  unsigned short* Kb = Qb + (size_t)BS * 2048;                    // [BS,512]  bf16
  unsigned short* Vb = Kb + (size_t)BS * 512;                     // [BS,512]  bf16
  unsigned short* AO = Vb + (size_t)BS * 512;                     // [BS,2048] bf16

  // QKV projections (bf16 WMMA, f32 accumulate)
  gemm_wmma_kernel<float><<<dim3(BS / 128, 2048 / 128), 256, 0, stream>>>(
      q_stream, wq, XQ, BS, 2048, 2048);
  gemm_wmma_kernel<float><<<dim3(BS / 128, 512 / 128), 256, 0, stream>>>(
      kv_stream, wk, XK, BS, 512, 2048);
  gemm_wmma_kernel<float><<<dim3(BS / 128, 512 / 128), 256, 0, stream>>>(
      kv_stream, wv, XV, BS, 512, 2048);

  // per-head RMSNorm (q, k) -> bf16; v -> bf16 cast
  rmsnorm_head_kernel<<<(BS * 16) / 8, 256, 0, stream>>>(XQ, q_norm_w, Qb, BS * 16);
  rmsnorm_head_kernel<<<(BS * 4) / 8, 256, 0, stream>>>(XK, k_norm_w, Kb, BS * 4);
  cast_bf16_kernel<<<1024, 256, 0, stream>>>(XV, Vb, BS * 512);

  // causal flash attention (async-to-LDS staged K/V)
  flash_attn_kernel<<<dim3(2048 / 64, 16, 2), 128, 0, stream>>>(Qb, Kb, Vb, AO);

  // output projection -> fp32 result (A tile staged via async-to-LDS)
  gemm_wmma_kernel<unsigned short><<<dim3(BS / 128, 2048 / 128), 256, 0, stream>>>(
      AO, wo, out, BS, 2048, 2048);
}
